// T_MatrixVLinear_45140106281384
// MI455X (gfx1250) — matrix operand order repaired and checked by reading, not compiled, not run on hardware
//
#include <hip/hip_runtime.h>
#include <math.h>

// Problem dims (fixed by the reference)
#define M_DIM   4096
#define K_DIM   4096
#define BS_DIM  8192                 // BATCH * SEQ
#define NTOT    (M_DIM * K_DIM)      // 16777216
#define NKT     (K_DIM / 32)         // 128 k-steps of 32
#define LDA     40                   // LDS row pitch in f16 (32 + 8 pad, keeps 16B align)

typedef __attribute__((ext_vector_type(16))) _Float16 v16h;
typedef __attribute__((ext_vector_type(8)))  _Float16 v8h;
typedef __attribute__((ext_vector_type(4)))  _Float16 v4h;
typedef __attribute__((ext_vector_type(8)))  float    v8f;

union ABu { v16h v; v8h h[2]; };

// Async DMA: global (per-lane 16B) -> LDS, tracked by ASYNCcnt.
#define ASYNC_CP_B128(ldsaddr, gptr)                                  \
  asm volatile("global_load_async_to_lds_b128 %0, %1, off"            \
               :: "v"(ldsaddr), "v"(gptr) : "memory")
#define WAIT_ASYNC0() asm volatile("s_wait_asynccnt 0" ::: "memory")

// Gielis superformula radius from a Hilbert index.
__device__ __forceinline__ float gielis_r(int d, float m_p, float a, float b,
                                          float inv_n1, float n2, float n3) {
  const float k = 6.283185307179586f / 16777216.0f;  // 2*pi / 2^(2*ORDER)
  float ang = m_p * ((float)d * k) * 0.25f;
  float cb  = powf(fabsf(cosf(ang) / a), n2);
  float sb  = powf(fabsf(sinf(ang) / b), n3);
  return powf(cb + sb, -inv_n1);
}

// ---------------- Stage 1: deterministic two-pass mean/std ----------------
__global__ __launch_bounds__(256) void k_stats_partial(
    const int* __restrict__ hd, const float* __restrict__ dna,
    double* __restrict__ psum, double* __restrict__ psq) {
  __shared__ double ss[256], sq[256];
  const float m_p = dna[0], a = dna[1], b = dna[2];
  const float inv_n1 = 1.0f / dna[3], n2 = dna[4], n3 = dna[5];
  double s = 0.0, q = 0.0;
  for (int i = blockIdx.x * 256 + threadIdx.x; i < NTOT; i += gridDim.x * 256) {
    float r = gielis_r(hd[i], m_p, a, b, inv_n1, n2, n3);
    s += (double)r; q += (double)r * (double)r;
  }
  ss[threadIdx.x] = s; sq[threadIdx.x] = q;
  __syncthreads();
  for (int off = 128; off > 0; off >>= 1) {
    if (threadIdx.x < off) {
      ss[threadIdx.x] += ss[threadIdx.x + off];
      sq[threadIdx.x] += sq[threadIdx.x + off];
    }
    __syncthreads();
  }
  if (threadIdx.x == 0) { psum[blockIdx.x] = ss[0]; psq[blockIdx.x] = sq[0]; }
}

__global__ __launch_bounds__(256) void k_stats_final(
    const double* __restrict__ psum, const double* __restrict__ psq,
    float* __restrict__ stats, int nparts) {
  __shared__ double ss[256], sq[256];
  double s = 0.0, q = 0.0;
  for (int i = threadIdx.x; i < nparts; i += 256) { s += psum[i]; q += psq[i]; }
  ss[threadIdx.x] = s; sq[threadIdx.x] = q;
  __syncthreads();
  for (int off = 128; off > 0; off >>= 1) {
    if (threadIdx.x < off) {
      ss[threadIdx.x] += ss[threadIdx.x + off];
      sq[threadIdx.x] += sq[threadIdx.x + off];
    }
    __syncthreads();
  }
  if (threadIdx.x == 0) {
    const double N   = (double)NTOT;
    const double mu  = ss[0] / N;
    const double var = (sq[0] - ss[0] * ss[0] / N) / (N - 1.0);  // unbiased
    const double sd  = sqrt(var);
    stats[0] = (float)mu;
    stats[1] = (float)((1.0 / 64.0) / (sd + 1e-9));  // fold scale = 1/sqrt(K)
  }
}

// ---------------- Stage 2: materialize w (f16) and x (f16) ----------------
__global__ __launch_bounds__(256) void k_build_w(
    const int* __restrict__ hd, const float* __restrict__ dna,
    const float* __restrict__ stats, _Float16* __restrict__ wh) {
  const int i = blockIdx.x * 256 + threadIdx.x;
  if (i >= NTOT) return;
  const float m_p = dna[0], a = dna[1], b = dna[2];
  const float inv_n1 = 1.0f / dna[3], n2 = dna[4], n3 = dna[5];
  const float mean = stats[0], coef = stats[1];
  float r = gielis_r(hd[i], m_p, a, b, inv_n1, n2, n3);
  wh[i] = (_Float16)((r - mean) * coef);
}

__global__ __launch_bounds__(256) void k_conv_x(
    const float* __restrict__ x, _Float16* __restrict__ xh) {
  const int i = (blockIdx.x * 256 + threadIdx.x) * 4;
  float4 v = *(const float4*)(x + i);
  v4h h;
  h.x = (_Float16)v.x; h.y = (_Float16)v.y;
  h.z = (_Float16)v.z; h.w = (_Float16)v.w;
  *(v4h*)(xh + i) = h;
}

// ---------------- Stage 3: WMMA GEMM  out[bs,m] = x·w^T + bias ----------------
// Block: 256 thr = 8 waves (wave32). Block tile 128(bs) x 128(m), K-step 32.
// Wave grid 2x4 -> each wave owns a 64x32 tile = 4x2 WMMA 16x16 accumulators.
// Tiles are DMA'd into double-buffered LDS with GLOBAL_LOAD_ASYNC_TO_LDS_B128;
// the k+1 DMA runs behind the 8 WMMAs of step k.
__global__ __launch_bounds__(256) void k_gemm(
    const _Float16* __restrict__ xh, const _Float16* __restrict__ wh,
    const float* __restrict__ bias, float* __restrict__ out) {
  __shared__ __align__(16) _Float16 ldsA[2][128 * LDA];
  __shared__ __align__(16) _Float16 ldsB[2][128 * LDA];

  const int tid   = threadIdx.x;
  const int lane  = tid & 31;
  const int wid   = tid >> 5;
  const int wr    = wid >> 2;     // 0..1  (64-row slab of bs)
  const int wc    = wid & 3;      // 0..3  (32-col slab of m)
  const int lrow  = lane & 15;
  const int khalf = lane >> 4;

  const int m0  = blockIdx.x * 128;
  const int bs0 = blockIdx.y * 128;

  // Cooperative tile DMA mapping: 512 chunks of 8 f16 per 128x32 tile,
  // each thread owns chunk rows ar0 and ar0+64 at k-offset ac.
  const int ar0 = tid >> 2;        // rows 0..63
  const int ar1 = ar0 + 64;        // rows 64..127
  const int ac  = (tid & 3) * 8;   // k-chunk within row

  const _Float16* aG0 = xh + (bs0 + ar0) * K_DIM + ac;
  const _Float16* aG1 = xh + (bs0 + ar1) * K_DIM + ac;
  const _Float16* bG0 = wh + (m0 + ar0) * K_DIM + ac;
  const _Float16* bG1 = wh + (m0 + ar1) * K_DIM + ac;

  // LDS byte addresses (generic LDS pointer low 32 bits == LDS offset).
  const unsigned c0    = (unsigned)((ar0 * LDA + ac) * 2);
  const unsigned c1    = (unsigned)((ar1 * LDA + ac) * 2);
  const unsigned aBase = (unsigned)(size_t)&ldsA[0][0];
  const unsigned bBase = (unsigned)(size_t)&ldsB[0][0];
  const unsigned BUFB  = 128 * LDA * 2;   // bytes per buffer

  auto asyncTile = [&](int kt, int buf) {
    const int ko = kt * 32;
    const unsigned o = (unsigned)buf * BUFB;
    ASYNC_CP_B128(aBase + o + c0, aG0 + ko);
    ASYNC_CP_B128(aBase + o + c1, aG1 + ko);
    ASYNC_CP_B128(bBase + o + c0, bG0 + ko);
    ASYNC_CP_B128(bBase + o + c1, bG1 + ko);
  };

  v8f acc[4][2];
#pragma unroll
  for (int i = 0; i < 4; ++i) {
#pragma unroll
    for (int j = 0; j < 2; ++j) {
#pragma unroll
      for (int e = 0; e < 8; ++e) acc[i][j][e] = 0.0f;
    }
  }

  asyncTile(0, 0);
  WAIT_ASYNC0();
  __syncthreads();

#pragma unroll 2
  for (int kt = 0; kt < NKT; ++kt) {
    const int cur = kt & 1;
    if (kt + 1 < NKT) asyncTile(kt + 1, cur ^ 1);  // DMA next tile behind WMMA

    // A fragment (16x32 f16): lane holds row (lane&15), K runs
    // [8*khalf, +8) in VGPR0-3 and [16+8*khalf, +8) in VGPR4-7.
    ABu af[4], bf[2];
#pragma unroll
    for (int i = 0; i < 4; ++i) {
      const _Float16* p = &ldsA[cur][(wr * 64 + i * 16 + lrow) * LDA];
      af[i].h[0] = *(const v8h*)(p + khalf * 8);
      af[i].h[1] = *(const v8h*)(p + 16 + khalf * 8);
    }
    // B fragment (32x16 f16): lane holds column (lane&15),
// K [8*khalf, +8) in VGPR0-3 and [16+8*khalf, +8) in VGPR4-7 (same as A).
#pragma unroll
    for (int j = 0; j < 2; ++j) {
      const _Float16* p = &ldsB[cur][(wc * 32 + j * 16 + lrow) * LDA];
      bf[j].h[0] = *(const v8h*)(p + khalf * 8);
      bf[j].h[1] = *(const v8h*)(p + 16 + khalf * 8);
    }

#pragma unroll
    for (int i = 0; i < 4; ++i) {
#pragma unroll
      for (int j = 0; j < 2; ++j) {
        acc[i][j] = __builtin_amdgcn_wmma_f32_16x16x32_f16(
            false, af[i].v, false, bf[j].v, (short)0, acc[i][j], false, false);
      }
    }

    WAIT_ASYNC0();      // this wave's DMA into the next buffer has landed
    __syncthreads();    // everyone's DMA visible before next step reads it
  }

  // C/D layout: VGPR r of lane L holds (row = r + 8*(L/16), col = L%16).
#pragma unroll
  for (int i = 0; i < 4; ++i) {
    const int row0 = bs0 + wr * 64 + i * 16 + khalf * 8;
#pragma unroll
    for (int j = 0; j < 2; ++j) {
      const int col = m0 + wc * 32 + j * 16 + lrow;
      const float bv = bias[col];
#pragma unroll
      for (int e = 0; e < 8; ++e)
        out[(row0 + e) * M_DIM + col] = acc[i][j][e] + bv;
    }
  }
}

// ---------------- host launcher ----------------
extern "C" void kernel_launch(void* const* d_in, const int* in_sizes, int n_in,
                              void* d_out, int out_size, void* d_ws, size_t ws_size,
                              hipStream_t stream) {
  const float* x    = (const float*)d_in[0];
  const float* dna  = (const float*)d_in[1];
  const float* bias = (const float*)d_in[2];
  const int*   hd   = (const int*)d_in[3];
  float* out = (float*)d_out;

  // Workspace layout: w_h (32 MiB) | x_h (64 MiB) | partials (64 KiB) | stats
  char* ws = (char*)d_ws;
  _Float16* wh = (_Float16*)ws;
  _Float16* xh = (_Float16*)(ws + (size_t)NTOT * 2);
  double* psum = (double*)(ws + (size_t)NTOT * 2 + (size_t)BS_DIM * K_DIM * 2);
  double* psq  = psum + 4096;
  float* stats = (float*)(psq + 4096);

  k_stats_partial<<<4096, 256, 0, stream>>>(hd, dna, psum, psq);
  k_stats_final<<<1, 256, 0, stream>>>(psum, psq, stats, 4096);
  k_build_w<<<NTOT / 256, 256, 0, stream>>>(hd, dna, stats, wh);
  k_conv_x<<<(BS_DIM * K_DIM) / (256 * 4), 256, 0, stream>>>(x, xh);
  dim3 g(M_DIM / 128, BS_DIM / 128);
  k_gemm<<<g, 256, 0, stream>>>(xh, wh, bias, out);
}
